// PositionAttention_26749056320171
// MI455X (gfx1250) — matrix operand order repaired and checked by reading, not compiled, not run on hardware
//
#include <hip/hip_runtime.h>
#include <hip/hip_bf16.h>

// PositionAttention for MI455X (gfx1250), flash-attention style, all GEMMs on
// v_wmma_f32_16x16x32_f16 (fp16 inputs, fp32 accumulate).
//
// Shapes: B=4, C=256, Cq=32, N=H*W=4096.
// Workspace layout (~10.2 MB of d_ws, f16):
//   qfrag: [B][N/16][32 lanes][16]   A-layout fragments of q (pre-scaled by
//                                    log2(e) so softmax uses exp2)     (1 MB)
//   kfrag: [B][N/16][32 lanes][16]   B-layout fragments of k           (1 MB)
//   vfrag: [B][N/32][16 c-tiles][32 lanes][16] B-layout of v           (8 MB)
//   wqf  : [2 o-tiles][8 kc][32][16] A-fragments of log2(e)*Wq        (16 KB)
//   wkf  : [2 o-tiles][8 kc][32][16] A-fragments of Wk                (16 KB)
//   wvf  : [16 c-tiles][8 kc][32][16] A-fragments of Wv              (128 KB)

typedef __attribute__((ext_vector_type(16))) _Float16 v16h;
typedef __attribute__((ext_vector_type(8)))  float    v8f;

#define BATCH 4
#define CDIM  256
#define CQ    32
#define NPIX  4096
#define NT    (NPIX / 16)   // 256 column tiles
#define JCNT  (NPIX / 32)   // 128 j-chunks
#define LOG2E 1.44269504088896340736f

// 16-bit A-matrix 16x32 K-index for vector element e (0..15), lane group g:
// contiguous, matching B: lanes 0-15 -> K0..15, lanes 16-31 -> K16..31.
__device__ __forceinline__ int frag_kA(int e, int g) {
  return 16 * g + e;
}
// 16-bit B-matrix 32x16, this code's k order (same as A): lanes 0-15 hold K=0..15, lanes 16-31 K=16..31,
// contiguous within a lane -> k = 16*g + e  (so a B fragment is one v16h load
// from any [col][k]-contiguous buffer).

__device__ __forceinline__ v8f wmma_f16(v16h a, v16h b, v8f c) {
  return __builtin_amdgcn_wmma_f32_16x16x32_f16(false, a, false, b,
                                                (short)0, c, false, false);
}

// ---------------------------------------------------------------------------
// Kernel 0: pre-pack weight matrices into WMMA A-fragments (done once; the
// same fragment is reused by every (batch, column-tile) GEMM block).
// Wq is pre-scaled by log2(e) so the attention softmax can use exp2 natively.
// blockIdx: 0..15 -> Wq tiles, 16..31 -> Wk tiles, 32..159 -> Wv tiles.
// ---------------------------------------------------------------------------
__global__ __launch_bounds__(32)
void pack_w_frags(const float* __restrict__ Wq, const float* __restrict__ Wk,
                  const float* __restrict__ Wv,
                  _Float16* __restrict__ wqf, _Float16* __restrict__ wkf,
                  _Float16* __restrict__ wvf) {
  int lane = threadIdx.x & 31, g = lane >> 4, ln = lane & 15;
  int idx = blockIdx.x;
  const float* W; _Float16* dst; float scale; int t;
  if (idx < 16)      { W = Wq; dst = wqf; t = idx;      scale = LOG2E; }
  else if (idx < 32) { W = Wk; dst = wkf; t = idx - 16; scale = 1.f;   }
  else               { W = Wv; dst = wvf; t = idx - 32; scale = 1.f;   }
  int o0 = (t >> 3) * 16, k0 = (t & 7) * 32;
  v16h a;
#pragma unroll
  for (int e = 0; e < 16; ++e)
    a[e] = (_Float16)(W[(size_t)(o0 + ln) * CDIM + k0 + frag_kA(e, g)] * scale);
  *(v16h*)(dst + ((size_t)t * 32 + lane) * 16) = a;
}

// ---------------------------------------------------------------------------
// Kernel 1: q,k projections -> pre-swizzled WMMA fragments.
// One block per (batch, 16-column tile). 4 waves: (q|k) x (d-rows 0-15|16-31).
// x tile staged TRANSPOSED in LDS so B fragments are contiguous v16h loads.
// ---------------------------------------------------------------------------
__global__ __launch_bounds__(128)
void proj_qk_kernel(const float* __restrict__ x,
                    const float* __restrict__ bq, const float* __restrict__ bk,
                    const _Float16* __restrict__ wqf,
                    const _Float16* __restrict__ wkf,
                    _Float16* __restrict__ qfrag, _Float16* __restrict__ kfrag) {
  __shared__ _Float16 ldsXT[16][CDIM];     // [n][c], 8 KB
  __shared__ _Float16 stage[2][32][16];    // [q|k][d][col], 2 KB

  int b = blockIdx.x / NT, nt = blockIdx.x % NT, n0 = nt * 16;
  int tid = threadIdx.x;
  const float* xb = x + ((size_t)b * CDIM) * NPIX + n0;
  for (int i = tid; i < CDIM * 16; i += 128) {
    int c = i >> 4, n = i & 15;
    ldsXT[n][c] = (_Float16)xb[(size_t)c * NPIX + n];
  }
  __syncthreads();

  int wave = tid >> 5, lane = tid & 31, g = lane >> 4, ln = lane & 15;
  int sel = wave >> 1;                 // 0 = q, 1 = k
  int o0 = (wave & 1) * 16;            // d-row offset
  const _Float16* wf   = sel ? wkf : wqf;
  const float*    bias = sel ? bk  : bq;
  float bsc = sel ? 1.f : LOG2E;       // bq folded into log2 domain like Wq

  v8f acc = {};
  for (int kc = 0; kc < 8; ++kc) {     // K = 256 in chunks of 32
    v16h a  = *(const v16h*)(wf + ((size_t)((wave & 1) * 8 + kc) * 32 + lane) * 16);
    v16h bf = *(const v16h*)&ldsXT[ln][kc * 32 + 16 * g];
    acc = wmma_f16(a, bf, acc);
  }
#pragma unroll
  for (int r = 0; r < 8; ++r) {        // D-layout: (m = r + 8g, n = ln)
    int m = r + 8 * g;
    stage[sel][o0 + m][ln] = (_Float16)(acc[r] + bias[o0 + m] * bsc);
  }
  __syncthreads();

  if (wave == 0) {                     // emit Q as A-fragment (lane = row i)
    v16h q;
#pragma unroll
    for (int e = 0; e < 16; ++e) q[e] = stage[0][frag_kA(e, g)][ln];
    *(v16h*)(qfrag + ((size_t)(b * NT + nt) * 32 + lane) * 16) = q;
  } else if (wave == 2) {              // emit K as B-fragment (lane = col j)
    v16h k;
#pragma unroll
    for (int e = 0; e < 16; ++e) k[e] = stage[1][16 * g + e][ln];
    *(v16h*)(kfrag + ((size_t)(b * NT + nt) * 32 + lane) * 16) = k;
  }
}

// ---------------------------------------------------------------------------
// Kernel 2: v projection -> B-fragments per (32-wide j-chunk, 16-wide c-tile).
// One block per (batch, j-chunk). Wave w handles c-tiles w, w+4, w+8, w+12.
// ---------------------------------------------------------------------------
__global__ __launch_bounds__(128)
void proj_v_kernel(const float* __restrict__ x, const float* __restrict__ bv,
                   const _Float16* __restrict__ wvf,
                   _Float16* __restrict__ vfrag) {
  __shared__ _Float16 ldsXT[32][CDIM];     // [j][c], 16 KB
  __shared__ _Float16 stageV[4][16][32];   // per-wave [c][j] staging, 4 KB

  int b = blockIdx.x / JCNT, jc = blockIdx.x % JCNT, j0 = jc * 32;
  int tid = threadIdx.x;
  const float* xb = x + ((size_t)b * CDIM) * NPIX + j0;
  for (int i = tid; i < CDIM * 32; i += 128) {
    int c = i >> 5, j = i & 31;
    ldsXT[j][c] = (_Float16)xb[(size_t)c * NPIX + j];
  }
  __syncthreads();

  int wave = tid >> 5, lane = tid & 31, g = lane >> 4, ln = lane & 15;
  for (int ci = 0; ci < 4; ++ci) {
    int ct = wave + 4 * ci, c0 = ct * 16;
    v8f acc0 = {}, acc1 = {};
    for (int kc = 0; kc < 8; ++kc) {
      v16h a  = *(const v16h*)(wvf + ((size_t)(ct * 8 + kc) * 32 + lane) * 16);
      v16h b0 = *(const v16h*)&ldsXT[ln][kc * 32 + 16 * g];       // j-tile 0
      v16h b1 = *(const v16h*)&ldsXT[16 + ln][kc * 32 + 16 * g];  // j-tile 1
      acc0 = wmma_f16(a, b0, acc0);
      acc1 = wmma_f16(a, b1, acc1);
    }
#pragma unroll
    for (int r = 0; r < 8; ++r) {
      int m = r + 8 * g;
      float bb = bv[c0 + m];
      stageV[wave][m][ln]      = (_Float16)(acc0[r] + bb);
      stageV[wave][m][16 + ln] = (_Float16)(acc1[r] + bb);
    }
    // same-wave LDS RAW: DS ops are in-order, compiler inserts s_wait_dscnt
    v16h vf = *(const v16h*)&stageV[wave][ln][16 * g];
    *(v16h*)(vfrag + (((size_t)(b * JCNT + jc) * 16 + ct) * 32 + lane) * 16) = vf;
    __builtin_amdgcn_wave_barrier();
  }
}

// ---------------------------------------------------------------------------
// Kernel 3: flash attention + gamma*out + x. One block per (batch, 16 queries).
// Wave w owns c in [64w, 64w+64): 4 f32 D-layout accumulators. Scores arrive
// pre-scaled by log2(e) (folded into Wq/bq), so softmax uses exp2 directly.
// Each wave redundantly computes the score tile: QK is 2 of 6 WMMAs/chunk and
// this keeps waves free of per-chunk __syncthreads.
// ---------------------------------------------------------------------------
__global__ __launch_bounds__(128)
void attn_kernel(const float* __restrict__ x, const float* __restrict__ gamma,
                 const _Float16* __restrict__ qfrag,
                 const _Float16* __restrict__ kfrag,
                 const _Float16* __restrict__ vfrag,
                 float* __restrict__ out) {
  __shared__ _Float16 stageP[4][16][32];   // per-wave P transpose, 4 KB

  int b = blockIdx.x / NT, it = blockIdx.x % NT, i0 = it * 16;
  int tid = threadIdx.x, wave = tid >> 5, lane = tid & 31;
  int g = lane >> 4, ln = lane & 15;

  v16h qa = *(const v16h*)(qfrag + ((size_t)(b * NT + it) * 32 + lane) * 16);
  float mrow[8], lrow[8];
  v8f acc[4] = {{}, {}, {}, {}};
#pragma unroll
  for (int r = 0; r < 8; ++r) { mrow[r] = -3.0e38f; lrow[r] = 0.f; }

  for (int jc = 0; jc < JCNT; ++jc) {
    const _Float16* kb = kfrag + ((size_t)(b * NT + 2 * jc) * 32 + lane) * 16;
    const _Float16* vb0 = vfrag +
        (((size_t)(b * JCNT + jc) * 16 + wave * 4) * 32 + lane) * 16;
    if (jc + 1 < JCNT) {               // global_prefetch_b8 of next chunk
      __builtin_prefetch(kb + 2 * 32 * 16, 0, 1);
      __builtin_prefetch(vb0 + (size_t)16 * 32 * 16, 0, 1);
    }
    v16h kf0 = *(const v16h*)(kb);
    v16h kf1 = *(const v16h*)(kb + 32 * 16);
    v8f z = {};
    v8f s0 = wmma_f16(qa, kf0, z);     // log2-domain scores, j-tile 2jc
    v8f s1 = wmma_f16(qa, kf1, z);     // log2-domain scores, j-tile 2jc+1

#pragma unroll
    for (int r = 0; r < 8; ++r) {
      float t = fmaxf(s0[r], s1[r]);               // row max over this chunk
      t = fmaxf(t, __shfl_xor(t, 1, 32));
      t = fmaxf(t, __shfl_xor(t, 2, 32));
      t = fmaxf(t, __shfl_xor(t, 4, 32));
      t = fmaxf(t, __shfl_xor(t, 8, 32));
      float newm  = fmaxf(mrow[r], t);
      float alpha = exp2f(mrow[r] - newm);         // one v_exp_f32 each
      float p0 = exp2f(s0[r] - newm);
      float p1 = exp2f(s1[r] - newm);
      float rs = p0 + p1;                          // row sum over this chunk
      rs += __shfl_xor(rs, 1, 32);
      rs += __shfl_xor(rs, 2, 32);
      rs += __shfl_xor(rs, 4, 32);
      rs += __shfl_xor(rs, 8, 32);
      lrow[r] = lrow[r] * alpha + rs;
      mrow[r] = newm;
#pragma unroll
      for (int t4 = 0; t4 < 4; ++t4) acc[t4][r] *= alpha;  // rescale O
      stageP[wave][r + 8 * g][ln]      = (_Float16)p0;     // D->plain [m][j]
      stageP[wave][r + 8 * g][16 + ln] = (_Float16)p1;
    }
    // same-wave LDS transpose read into A-layout (lane = row m)
    v16h pa;
#pragma unroll
    for (int e = 0; e < 16; ++e) pa[e] = stageP[wave][ln][frag_kA(e, g)];
#pragma unroll
    for (int t4 = 0; t4 < 4; ++t4) {
      v16h vb = *(const v16h*)(vb0 + (size_t)t4 * 32 * 16);
      acc[t4] = wmma_f16(pa, vb, acc[t4]);         // O += P x V
    }
    __builtin_amdgcn_wave_barrier();               // before stageP reuse (WAR)
  }

  float gm = gamma[0];
  float inv[8];
#pragma unroll
  for (int r = 0; r < 8; ++r) inv[r] = 1.0f / lrow[r];  // 8 divides, not 32
#pragma unroll
  for (int t4 = 0; t4 < 4; ++t4) {
    int c = (wave * 4 + t4) * 16 + ln;             // lane's channel
    const float* xp = x   + ((size_t)b * CDIM + c) * NPIX + i0 + 8 * g;
    float*       op = out + ((size_t)b * CDIM + c) * NPIX + i0 + 8 * g;
#pragma unroll
    for (int r = 0; r < 8; ++r)                    // 8 consecutive pixels
      op[r] = gm * (acc[t4][r] * inv[r]) + xp[r];
  }
}

// ---------------------------------------------------------------------------
extern "C" void kernel_launch(void* const* d_in, const int* in_sizes, int n_in,
                              void* d_out, int out_size, void* d_ws, size_t ws_size,
                              hipStream_t stream) {
  const float* x     = (const float*)d_in[0];
  const float* Wq    = (const float*)d_in[1];
  const float* bq    = (const float*)d_in[2];
  const float* Wk    = (const float*)d_in[3];
  const float* bk    = (const float*)d_in[4];
  const float* Wv    = (const float*)d_in[5];
  const float* bv    = (const float*)d_in[6];
  const float* gamma = (const float*)d_in[7];
  float* out = (float*)d_out;

  _Float16* qfrag = (_Float16*)d_ws;                                // 1 MB
  _Float16* kfrag = qfrag + (size_t)BATCH * NT * 32 * 16;           // 1 MB
  _Float16* vfrag = kfrag + (size_t)BATCH * NT * 32 * 16;           // 8 MB
  _Float16* wqf   = vfrag + (size_t)BATCH * JCNT * 16 * 32 * 16;    // 16 KB
  _Float16* wkf   = wqf + (size_t)16 * 32 * 16;                     // 16 KB
  _Float16* wvf   = wkf + (size_t)16 * 32 * 16;                     // 128 KB
  (void)ws_size; (void)in_sizes; (void)n_in; (void)out_size;

  pack_w_frags<<<160, 32, 0, stream>>>(Wq, Wk, Wv, wqf, wkf, wvf);
  proj_qk_kernel<<<BATCH * NT, 128, 0, stream>>>(x, bq, bk, wqf, wkf, qfrag, kfrag);
  proj_v_kernel<<<BATCH * JCNT, 128, 0, stream>>>(x, bv, wvf, vfrag);
  attn_kernel<<<BATCH * NT, 128, 0, stream>>>(x, gamma, qfrag, kfrag, vfrag, out);
}
